// AttentionLSTM_82240033784243
// MI455X (gfx1250) — matrix operand order repaired and checked by reading, not compiled, not run on hardware
//
#include <hip/hip_runtime.h>
#include <hip/hip_bf16.h>
#include <stdint.h>

// Problem constants (from reference): N=256, T=128, D=512, H=1024
#define N_  256
#define T_  128
#define D_  512
#define H_  1024
#define FH_ 4096           // 4*H
#define K_  2560           // D + H + H  (fused GEMM K)
#define KITER_ (K_ / 32)   // 80 K-iterations

typedef __attribute__((ext_vector_type(16))) __bf16   v16bf;
typedef __attribute__((ext_vector_type(8)))  float    v8f;
typedef __attribute__((ext_vector_type(4)))  float    f32x4;
typedef __attribute__((ext_vector_type(4)))  uint32_t u32x4;

union Frag { u32x4 u4[2]; v16bf v; };   // 32 bytes: one bf16 WMMA A/B fragment per lane

__device__ __forceinline__ float sigf(float x) { return 1.0f / (1.0f + __expf(-x)); }

// ---------------------------------------------------------------------------
// One-time: fuse Wx(512x4096), Wh(1024x4096), Wattn(1024x4096) into an
// N-major bf16 matrix Wt[n][k] (4096 x 2560) so GEMM B-tiles are contiguous.
// ---------------------------------------------------------------------------
__global__ __launch_bounds__(256) void convert_weights(
    const float* __restrict__ Wx, const float* __restrict__ Wh,
    const float* __restrict__ Wattn, __bf16* __restrict__ Wt)
{
  int id = blockIdx.x * 256 + threadIdx.x;   // id = k*4096 + n  (contiguous reads)
  int nn = id & (FH_ - 1);
  int k  = id >> 12;
  float v;
  if (k < D_)             v = Wx[(size_t)k * FH_ + nn];
  else if (k < D_ + H_)   v = Wh[(size_t)(k - D_) * FH_ + nn];
  else                    v = Wattn[(size_t)(k - D_ - H_) * FH_ + nn];
  Wt[(size_t)nn * K_ + k] = (__bf16)v;
}

// h0 = c0 = mean of A over 4x4 grid; also seed bf16 h panel in Apack.
__global__ __launch_bounds__(256) void init_state(
    const float* __restrict__ A, float* __restrict__ hbuf, float* __restrict__ cbuf,
    __bf16* __restrict__ Apack)
{
  int id = blockIdx.x * 256 + threadIdx.x;   // n*H + h
  int n = id >> 10, h = id & (H_ - 1);
  const f32x4* a = (const f32x4*)(A + (size_t)id * 16);
  float s = 0.f;
  #pragma unroll
  for (int q = 0; q < 4; ++q) { f32x4 v = a[q]; s += v.x + v.y + v.z + v.w; }
  s *= 0.0625f;
  hbuf[id] = s; cbuf[id] = s;
  Apack[(size_t)n * K_ + D_ + h] = (__bf16)s;
}

// Per step: pack xt into Apack[:, 0:512] as bf16 (4 elements per thread).
__global__ __launch_bounds__(256) void pack_x(
    const float* __restrict__ x, __bf16* __restrict__ Apack, int tstep)
{
  int id = (blockIdx.x * 256 + threadIdx.x) * 4;   // n*512 + d
  int n = id >> 9, d = id & (D_ - 1);
  f32x4 v = *(const f32x4*)(x + ((size_t)n * T_ + tstep) * D_ + d);
  __bf16* dst = Apack + (size_t)n * K_ + d;
  dst[0] = (__bf16)v.x; dst[1] = (__bf16)v.y; dst[2] = (__bf16)v.z; dst[3] = (__bf16)v.w;
}

// ---------------------------------------------------------------------------
// Attention: per sample n, M_k = (1/sqrt(H)) * sum_h h[n,h]*A2[n,h,k],
// w = softmax(M), attn[n,h] = sum_k A2[n,h,k]*w_k -> bf16 into Apack.
// ---------------------------------------------------------------------------
__global__ __launch_bounds__(256) void attn_step(
    const float* __restrict__ A2, const float* __restrict__ hbuf,
    __bf16* __restrict__ Apack)
{
  const int n = blockIdx.x;
  const int tid = threadIdx.x, lane = tid & 31, wave = tid >> 5;
  __shared__ float red[8][16];
  __shared__ float logits[16];

  float m[16];
  #pragma unroll
  for (int k = 0; k < 16; ++k) m[k] = 0.f;

  for (int h = tid; h < H_; h += 256) {
    float hv = hbuf[(size_t)n * H_ + h];
    const f32x4* a4 = (const f32x4*)(A2 + ((size_t)n * H_ + h) * 16);
    #pragma unroll
    for (int q = 0; q < 4; ++q) {
      f32x4 a = a4[q];
      m[q*4+0] += hv * a.x; m[q*4+1] += hv * a.y;
      m[q*4+2] += hv * a.z; m[q*4+3] += hv * a.w;
    }
  }
  // wave32 butterfly reduction of all 16 partials
  #pragma unroll
  for (int off = 16; off >= 1; off >>= 1)
    #pragma unroll
    for (int k = 0; k < 16; ++k) m[k] += __shfl_xor(m[k], off);
  if (lane == 0) {
    #pragma unroll
    for (int k = 0; k < 16; ++k) red[wave][k] = m[k];
  }
  __syncthreads();
  if (tid < 16) {
    float s = 0.f;
    #pragma unroll
    for (int w = 0; w < 8; ++w) s += red[w][tid];
    logits[tid] = s * 0.03125f;            // 1/sqrt(1024)
  }
  __syncthreads();

  float w[16], mx = logits[0];
  #pragma unroll
  for (int k = 1; k < 16; ++k) mx = fmaxf(mx, logits[k]);
  float sum = 0.f;
  #pragma unroll
  for (int k = 0; k < 16; ++k) { w[k] = __expf(logits[k] - mx); sum += w[k]; }
  float inv = 1.f / sum;

  for (int h = tid; h < H_; h += 256) {
    const f32x4* a4 = (const f32x4*)(A2 + ((size_t)n * H_ + h) * 16);
    float acc = 0.f;
    #pragma unroll
    for (int q = 0; q < 4; ++q) {
      f32x4 a = a4[q];
      acc += a.x * w[q*4+0] + a.y * w[q*4+1] + a.z * w[q*4+2] + a.w * w[q*4+3];
    }
    Apack[(size_t)n * K_ + (D_ + H_) + h] = (__bf16)(acc * inv);
  }
}

// ---------------------------------------------------------------------------
// Fused recurrent GEMM: act(256 x 4096) = Apack(256 x 2560, bf16) @ Wt^T.
// bf16 WMMA, f32 accumulate. 8 waves/block, wave tile 32x32, WG tile 64x128.
// Double-buffered LDS (compile-time buffer index), branch-free single-source
// staging: global_load_b128 -> ds_store_b128 only.
// ---------------------------------------------------------------------------
__global__ __launch_bounds__(256) void gemm_step(
    const __bf16* __restrict__ Apack, const __bf16* __restrict__ Wt,
    float* __restrict__ act)
{
  __shared__ __align__(16) __bf16 Alds[2][64][32];     // 2 x 4 KB  (M x K, row-major)
  __shared__ __align__(16) __bf16 Blds[2][128][32];    // 2 x 8 KB  (N x K, N-major)

  const int tid  = threadIdx.x;
  const int lane = tid & 31;
  const int wave = tid >> 5;
  const int wm = wave >> 2, wn = wave & 3;          // 2 x 4 wave grid
  const int mBlock = blockIdx.y, nBlock = blockIdx.x;

  // WMMA lane layout helpers (CDNA5 bf16 A 16x32 / B 32x16)
  const int hi  = lane >> 4;          // half-wave select
  const int lr  = lane & 15;          // row (A) / col (B) within fragment
  const int akb = hi ? 8 : 0;         // A: K base 0 or 8 (plus +16 for upper half)
  const int bkb = hi ? 8 : 0;         // B: K base 0 or 8 (plus +16 for upper half)

  // stage-load coordinates
  const int ar  = tid >> 2;           // A tile row 0..63
  const int akq = (tid & 3) * 8;      // A tile K offset
  const int bn  = tid >> 1;           // B tile row (output col) 0..127
  const int bkq = (tid & 1) * 16;
  const __bf16* asrc_row = Apack + (size_t)(mBlock * 64 + ar) * K_ + akq;
  const __bf16* bsrc_row = Wt + (size_t)(nBlock * 128 + bn) * K_ + bkq;

  auto stage = [&](int k0, int buf) {
    u32x4 a0 = *(const u32x4*)(asrc_row + k0);           // 8 bf16
    u32x4 b0 = *(const u32x4*)(bsrc_row + k0);           // 16 bf16
    u32x4 b1 = *(const u32x4*)(bsrc_row + k0 + 8);
    *(u32x4*)&Alds[buf][ar][akq]     = a0;
    *(u32x4*)&Blds[buf][bn][bkq]     = b0;
    *(u32x4*)&Blds[buf][bn][bkq + 8] = b1;
  };

  v8f acc[2][2];
  #pragma unroll
  for (int i = 0; i < 2; ++i)
    #pragma unroll
    for (int j = 0; j < 2; ++j) acc[i][j] = v8f{};

  auto compute = [&](int buf) {
    Frag a[2], b[2];
    #pragma unroll
    for (int mi = 0; mi < 2; ++mi) {
      int row = wm * 32 + mi * 16 + lr;
      a[mi].u4[0] = *(const u32x4*)&Alds[buf][row][akb];        // K = akb..akb+7
      a[mi].u4[1] = *(const u32x4*)&Alds[buf][row][akb + 16];   // K = 16+akb..
    }
    #pragma unroll
    for (int ni = 0; ni < 2; ++ni) {
      int row = wn * 32 + ni * 16 + lr;
      b[ni].u4[0] = *(const u32x4*)&Blds[buf][row][bkb];        // K = bkb..bkb+7
      b[ni].u4[1] = *(const u32x4*)&Blds[buf][row][bkb + 16];   // K = 16+bkb..
    }
    #pragma unroll
    for (int mi = 0; mi < 2; ++mi)
      #pragma unroll
      for (int ni = 0; ni < 2; ++ni)
        acc[mi][ni] = __builtin_amdgcn_wmma_f32_16x16x32_bf16(
            false, a[mi].v, false, b[ni].v, (short)0, acc[mi][ni], false, false);
  };

  stage(0, 0);
  __syncthreads();

  for (int it = 0; it < KITER_; it += 2) {
    const int k0 = it * 32;
    // even tile: compute buf0 while staging buf1
    stage(k0 + 32, 1);
    // Prefetch weight stream ~8 K-iterations ahead.  locality=3 -> WGP-scope
    // prefetch (fills all cache levels); the weights are L2-resident, so this
    // pulls them from GL2 into the WGP caches before the staging loads hit.
    {
      int pk = k0 + 256; if (pk > K_ - 32) pk = K_ - 32;
      __builtin_prefetch((const void*)(bsrc_row + pk), 0, 3);
    }
    compute(0);
    __syncthreads();
    // odd tile: compute buf1 while staging buf0 (skip stage past the end)
    if (it + 2 < KITER_) stage(k0 + 64, 0);
    compute(1);
    __syncthreads();
  }

  // ---- store f32 act; C/D layout: VGPR v, lane l -> M = v + 8*hi, N = l&15 ----
  #pragma unroll
  for (int mi = 0; mi < 2; ++mi) {
    #pragma unroll
    for (int ni = 0; ni < 2; ++ni) {
      int col = nBlock * 128 + wn * 32 + ni * 16 + lr;
      #pragma unroll
      for (int v = 0; v < 8; ++v) {
        int row = mBlock * 64 + wm * 32 + mi * 16 + v + hi * 8;
        act[(size_t)row * FH_ + col] = acc[mi][ni][v];
      }
    }
  }
}

// ---------------------------------------------------------------------------
// LSTM gates + state update + output write for timestep t.
// Writes next h as fp32 (for attention) and bf16 (into Apack for the GEMM).
// ---------------------------------------------------------------------------
__global__ __launch_bounds__(256) void lstm_pointwise(
    const float* __restrict__ act, const float* __restrict__ bias,
    float* __restrict__ hbuf, float* __restrict__ cbuf,
    __bf16* __restrict__ Apack, float* __restrict__ out, int tstep)
{
  int id = blockIdx.x * 256 + threadIdx.x;   // n*H + h
  int n = id >> 10, h = id & (H_ - 1);
  const float* ar = act + (size_t)n * FH_;
  float iv = sigf(ar[h]          + bias[h]);
  float fv = sigf(ar[H_ + h]     + bias[H_ + h]);
  float ov = sigf(ar[2*H_ + h]   + bias[2*H_ + h]);
  float gv = tanhf(ar[3*H_ + h]  + bias[3*H_ + h]);
  float c  = fv * cbuf[id] + iv * gv;
  float hn = ov * tanhf(c);
  cbuf[id] = c;
  hbuf[id] = hn;
  Apack[(size_t)n * K_ + D_ + h] = (__bf16)hn;
  out[((size_t)n * T_ + tstep) * H_ + h] = hn;
}

// ---------------------------------------------------------------------------
extern "C" void kernel_launch(void* const* d_in, const int* in_sizes, int n_in,
                              void* d_out, int out_size, void* d_ws, size_t ws_size,
                              hipStream_t stream) {
  const float* x     = (const float*)d_in[0];   // (N,T,D)
  const float* A     = (const float*)d_in[1];   // (N,H,4,4) == (N,H,16)
  const float* Wx    = (const float*)d_in[2];   // (D,4H)
  const float* Wh    = (const float*)d_in[3];   // (H,4H)
  const float* Wattn = (const float*)d_in[4];   // (H,4H)
  const float* bias  = (const float*)d_in[5];   // (4H)
  float* out = (float*)d_out;                   // (N,T,H)

  // workspace layout (~28 MB)
  char* ws = (char*)d_ws;
  __bf16* Wt    = (__bf16*)ws; ws += (size_t)FH_ * K_ * sizeof(__bf16);  // 20 MB
  __bf16* Apack = (__bf16*)ws; ws += (size_t)N_ * K_ * sizeof(__bf16);   // 1.25 MB
  float* hbuf   = (float*)ws;  ws += (size_t)N_ * H_ * sizeof(float);    // 1 MB
  float* cbuf   = (float*)ws;  ws += (size_t)N_ * H_ * sizeof(float);    // 1 MB
  float* act    = (float*)ws;                                            // 4 MB

  convert_weights<<<(FH_ * K_) / 256, 256, 0, stream>>>(Wx, Wh, Wattn, Wt);
  init_state<<<(N_ * H_) / 256, 256, 0, stream>>>(A, hbuf, cbuf, Apack);

  for (int t = 0; t < T_; ++t) {
    pack_x<<<(N_ * D_) / 1024, 256, 0, stream>>>(x, Apack, t);
    attn_step<<<N_, 256, 0, stream>>>(A, hbuf, Apack);
    gemm_step<<<dim3(FH_ / 128, N_ / 64), 256, 0, stream>>>(Apack, Wt, act);
    lstm_pointwise<<<(N_ * H_) / 256, 256, 0, stream>>>(act, bias, hbuf, cbuf,
                                                        Apack, out, t);
  }
}
